// ACModel_10084583211522
// MI455X (gfx1250) — matrix operand order repaired and checked by reading, not compiled, not run on hardware
//
#include <hip/hip_runtime.h>
#include <hip/hip_bf16.h>
#include <math.h>

// ---------------- CDNA5 WMMA types ----------------
typedef __attribute__((ext_vector_type(16))) _Float16 v16h;
typedef __attribute__((ext_vector_type(8)))  float    v8f;

#define BATCH 8192
#define NMEM  50
#define NTASK 7
#define NACT  7

// ---- LDS layout (float units) for main kernel (64 rows / block) ----
// [0, 9408)        image (NCHW, stride 147) -> conv2 (64x128) -> tmp1/tmp2
// [9408, 18624)    pooled (64x144)          -> embA (f16 A-frag, 2048 f)
#define OF_POOL   9408
#define OF_EMBA   9408      // 4096 halves = 2048 floats, 32B aligned (37632 B)
#define OF_TMP1   0         // 64x68 f32
#define OF_TMP2   4352
#define OF_EMBHA  11456     // 4096 halves = 2048 floats (45824 B, 32B aligned)
#define OF_QRY    13504     // 64x66 f32 -> 17728
#define OF_TIDS   18624     // after pool region (pool alive phases 1-2)
#define SM_MAIN_FLOATS 18688   // 74,752 bytes

// ---- LDS layout for memory kernel (50 rows) ----
// [0,7350) img / conv2 alias / tmp1 alias ; [7350,14550) pool ; embA at 7352
#define MF_EMBA   7352      // 29,408 B -> 32B aligned
#define SM_MEM_FLOATS 14550 // 58,200 bytes

// ---- workspace layout (bytes); matrices stored in B-fragment order ----
#define WS_MT1H  0                       // 7*4096 f16 (B-frag layout)
#define WS_MT2H  57344                   // 7*4096 f16 (B-frag layout)
#define WS_BT1   114688                  // 7*64 f32
#define WS_BT2   116480                  // 7*64 f32
#define WS_AW1H  118272                  // 64*64 f16 (B-frag layout)
#define WS_CW1H  126464                  // 64*64 f16 (B-frag layout)
#define WS_KEYV  134656                  // 50*64 f32
// total 147,456 bytes

// ------------- fragment index math (ISA 7.12.2 layouts, 64x64 tiles) -------------
// A fragment (16x32 f16): frag f = mt*2+ks (mt=row-tile, ks=k-tile); lane holds 16
// contiguous halves. kk=k&31: lane = (r&15) | ((kk>>3)&1)<<4 ; j = (kk&7)|((kk>>1)&8)
__device__ __forceinline__ int afrag_index(int r, int k) {
  int mt = r >> 4, ks = k >> 5, kk = k & 31;
  int lane = (r & 15) | (((kk >> 3) & 1) << 4);
  int j = (kk & 7) | ((kk >> 1) & 8);
  return ((mt * 2 + ks) * 32 + lane) * 16 + j;
}

// B fragment (32x16 f16): frag f = ks*4+nt; lane = ((kk>>3)&1)<<4 | (n&15); j = (kk&7)|((kk>>1)&8)
__device__ __forceinline__ int bfrag_index(int k, int n) {
  int frag = ((k >> 5) << 2) | (n >> 4);
  int kk = k & 31;
  int lane = (((kk >> 3) & 1) << 4) | (n & 15);
  return (frag * 32 + lane) * 16 + ((kk & 7) | ((kk >> 1) & 8));
}

// C/D (16x16 f32): VGPR v -> row m = v + 8*(lane>=16), col n = lane&15
__device__ __forceinline__ void store_c(float* dst, int ld, int row0, int n0, int lane, v8f c) {
  int n = n0 + (lane & 15);
  int mb = row0 + ((lane >> 4) << 3);
#pragma unroll
  for (int v = 0; v < 8; ++v) dst[(mb + v) * ld + n] = c[v];
}

// ---------------- fused conv pipeline (7x7x3 -> 64), emb emitted in A-frag f16 ----------------
__device__ void conv_pipeline(float* imgS, float* poolS, float* c2S, _Float16* embA, int nrows,
                              const float* w1, const float* b1,
                              const float* w2, const float* b2,
                              const float* w3, const float* b3, int tid) {
  // conv1(2x2,3->16) + relu + maxpool(2x2): 6x6 -> 3x3
  for (int p = tid; p < nrows * 16; p += 256) {
    int r = p >> 4, c = p & 15;
    float wr[12];
#pragma unroll
    for (int j = 0; j < 12; ++j) wr[j] = w1[c * 12 + j];
    float bias = b1[c];
    const float* im = imgS + r * 147;
    for (int py = 0; py < 3; ++py)
      for (int px = 0; px < 3; ++px) {
        float mx = 0.0f;                 // relu(.) >= 0
#pragma unroll
        for (int dy = 0; dy < 2; ++dy)
#pragma unroll
          for (int dx = 0; dx < 2; ++dx) {
            int y = 2 * py + dy, x = 2 * px + dx;
            float s = bias;
#pragma unroll
            for (int ic = 0; ic < 3; ++ic)
#pragma unroll
              for (int ky = 0; ky < 2; ++ky)
#pragma unroll
                for (int kx = 0; kx < 2; ++kx)
                  s += im[ic * 49 + (y + ky) * 7 + (x + kx)] * wr[ic * 4 + ky * 2 + kx];
            mx = fmaxf(mx, fmaxf(s, 0.0f));
          }
        poolS[r * 144 + c * 9 + py * 3 + px] = mx;
      }
  }
  __syncthreads();
  // conv2(2x2,16->32) + relu: 3x3 -> 2x2
  for (int p = tid; p < nrows * 32; p += 256) {
    int r = p >> 5, oc = p & 31;
    const float* pw = w2 + oc * 64;
    const float* pr = poolS + r * 144;
    float bias = b2[oc];
    for (int py = 0; py < 2; ++py)
      for (int px = 0; px < 2; ++px) {
        float s = bias;
        for (int ic = 0; ic < 16; ++ic)
#pragma unroll
          for (int ky = 0; ky < 2; ++ky)
#pragma unroll
            for (int kx = 0; kx < 2; ++kx)
              s += pr[ic * 9 + (py + ky) * 3 + (px + kx)] * pw[ic * 4 + ky * 2 + kx];
        c2S[r * 128 + oc * 4 + py * 2 + px] = fmaxf(s, 0.0f);
      }
  }
  __syncthreads();
  // zero the A-frag emb buffer (covers padded rows >= nrows)
  for (int p = tid; p < 2048; p += 256) ((float*)embA)[p] = 0.0f;
  __syncthreads();
  // conv3(2x2,32->64) + relu: 2x2 -> 1x1, scatter into A-frag layout
  for (int p = tid; p < nrows * 64; p += 256) {
    int r = p >> 6, oc = p & 63;
    const float* pw = w3 + oc * 128;
    const float* cr = c2S + r * 128;
    float s = b3[oc];
    for (int i = 0; i < 128; ++i) s += cr[i] * pw[i];
    embA[afrag_index(r, oc)] = (_Float16)fmaxf(s, 0.0f);
  }
  __syncthreads();
}

// ---------------- kernel 1: per-task hypernet matrices (B-frag layout) ----------------
__global__ __launch_bounds__(256) void acm_precompute(
    const float* __restrict__ te1, const float* __restrict__ te2,
    const float* __restrict__ Wm, const float* __restrict__ bm,
    const float* __restrict__ Wb, const float* __restrict__ bb0,
    const float* __restrict__ aw1, const float* __restrict__ cw1,
    _Float16* __restrict__ MT1H, _Float16* __restrict__ MT2H,
    float* __restrict__ BT1, float* __restrict__ BT2,
    _Float16* __restrict__ AW1H, _Float16* __restrict__ CW1H) {
  int t = blockIdx.x;
  int tid = threadIdx.x;
  if (t < NTASK) {
    float z1[8], z2[8];
#pragma unroll
    for (int i = 0; i < 8; ++i) { z1[i] = te1[t * 8 + i]; z2[i] = te2[t * 8 + i]; }
    for (int idx = tid; idx < 4096; idx += 256) {
      float a1 = bm[idx], a2 = bm[idx];
#pragma unroll
      for (int zi = 0; zi < 8; ++zi) {
        float w = Wm[zi * 4096 + idx];
        a1 += z1[zi] * w; a2 += z2[zi] * w;
      }
      int k = idx >> 6, n = idx & 63;           // M[d][e] -> B[k][n]
      int bo = t * 4096 + bfrag_index(k, n);
      MT1H[bo] = (_Float16)a1;
      MT2H[bo] = (_Float16)a2;
    }
    if (tid < 64) {
      float u1 = bb0[tid], u2 = bb0[tid];
#pragma unroll
      for (int zi = 0; zi < 8; ++zi) {
        float w = Wb[zi * 64 + tid];
        u1 += z1[zi] * w; u2 += z2[zi] * w;
      }
      BT1[t * 64 + tid] = u1;
      BT2[t * 64 + tid] = u2;
    }
  } else {
    for (int idx = tid; idx < 4096; idx += 256) {
      int bo = bfrag_index(idx >> 6, idx & 63);
      AW1H[bo] = (_Float16)aw1[idx];
      CW1H[bo] = (_Float16)cw1[idx];
    }
  }
}

// ---------------- kernel 2: memory bank embeds + keys (WMMA, rows padded to 64) ----------------
__global__ __launch_bounds__(256) void acm_memory(
    const float* __restrict__ mem_state, const int* __restrict__ mem_tids,
    const float* __restrict__ c1w, const float* __restrict__ c1b,
    const float* __restrict__ c2w, const float* __restrict__ c2b,
    const float* __restrict__ c3w, const float* __restrict__ c3b,
    const _Float16* __restrict__ MT2H, const float* __restrict__ BT2,
    float* __restrict__ KEYV) {
  extern __shared__ float sm[];
  int tid = threadIdx.x;
  int lane = tid & 31;
  int wave = tid >> 5;
  for (int p = tid; p < NMEM * 147; p += 256) sm[p] = mem_state[p];  // already NCHW
  __syncthreads();
  _Float16* embA = (_Float16*)(sm + MF_EMBA);
  conv_pipeline(sm, sm + 7350, sm, embA, NMEM, c1w, c1b, c2w, c2b, c3w, c3b, tid);

  float* tmp1 = sm;                                 // 64x68, conv2 region dead
  const v16h* embAF = (const v16h*)embA;

  int mi[2], ni[2];
  v16h aT[2][2];
#pragma unroll
  for (int t2 = 0; t2 < 2; ++t2) {
    int tIdx = wave * 2 + t2;
    mi[t2] = tIdx >> 2; ni[t2] = tIdx & 3;
#pragma unroll
    for (int ks = 0; ks < 2; ++ks)
      aT[t2][ks] = embAF[(mi[t2] * 2 + ks) * 32 + lane];
  }
  for (int t = 0; t < NTASK; ++t) {
    const v16h* B2 = (const v16h*)(MT2H + t * 4096);
#pragma unroll
    for (int t2 = 0; t2 < 2; ++t2) {
      v8f c = {};
#pragma unroll
      for (int ks = 0; ks < 2; ++ks)
        c = __builtin_amdgcn_wmma_f32_16x16x32_f16(
            false, aT[t2][ks], false, B2[((ks << 2) | ni[t2]) * 32 + lane],
            (short)0, c, false, false);
      store_c(tmp1, 68, mi[t2] * 16, ni[t2] * 16, lane, c);
    }
    __syncthreads();
    for (int p = tid; p < NMEM * 64; p += 256) {
      int n = p >> 6, e = p & 63;
      if (mem_tids[n] == t)
        KEYV[n * 64 + e] = tmp1[n * 68 + e] + BT2[t * 64 + e];
    }
    __syncthreads();
  }
}

// ---------------- kernel 3: fused batch forward (64 rows / block) ----------------
__global__ __launch_bounds__(256) void acm_main(
    const float* __restrict__ image, const int* __restrict__ text_ids,
    const float* __restrict__ mem_value,
    const float* __restrict__ c1w, const float* __restrict__ c1b,
    const float* __restrict__ c2w, const float* __restrict__ c2b,
    const float* __restrict__ c3w, const float* __restrict__ c3b,
    const float* __restrict__ aw2, const float* __restrict__ ab2,
    const float* __restrict__ ab1, const float* __restrict__ cb1,
    const float* __restrict__ cw2, const float* __restrict__ cb2,
    const _Float16* __restrict__ MT1H, const _Float16* __restrict__ MT2H,
    const float* __restrict__ BT1, const float* __restrict__ BT2,
    const _Float16* __restrict__ AW1H, const _Float16* __restrict__ CW1H,
    const float* __restrict__ KEYV, float* __restrict__ out) {
  extern __shared__ float sm[];
  int tid = threadIdx.x;
  int lane = tid & 31;
  int wave = tid >> 5;
  int row0 = blockIdx.x * 64;

  __builtin_prefetch(image + (size_t)row0 * 147, 0, 0);   // global_prefetch_b8

  // NHWC -> NCHW into LDS
  for (int p = tid; p < 64 * 147; p += 256) {
    int r = p / 147, j = p - r * 147;
    int c = j / 49, rem = j - c * 49;
    int h = rem / 7, w = rem - h * 7;
    sm[r * 147 + j] = image[(size_t)(row0 + r) * 147 + h * 21 + w * 3 + c];
  }
  if (tid < 64) ((int*)(sm + OF_TIDS))[tid] = text_ids[row0 + tid];
  __syncthreads();

  _Float16* embA = (_Float16*)(sm + OF_EMBA);
  conv_pipeline(sm, sm + OF_POOL, sm, embA, 64, c1w, c1b, c2w, c2b, c3w, c3b, tid);

  float*     tmp1  = sm + OF_TMP1;
  float*     tmp2  = sm + OF_TMP2;
  _Float16*  embhA = (_Float16*)(sm + OF_EMBHA);
  float*     qry   = sm + OF_QRY;
  const int* tidsS = (const int*)(sm + OF_TIDS);
  const v16h* embAF = (const v16h*)embA;

  // each wave owns 2 of the 16 output 16x16 tiles of the 64x64 result
  int mi[2], ni[2];
  v16h aT[2][2];
#pragma unroll
  for (int t2 = 0; t2 < 2; ++t2) {
    int tIdx = wave * 2 + t2;
    mi[t2] = tIdx >> 2; ni[t2] = tIdx & 3;
#pragma unroll
    for (int ks = 0; ks < 2; ++ks)
      aT[t2][ks] = embAF[(mi[t2] * 2 + ks) * 32 + lane];
  }

  // hypernet passes: 7 distinct matrices -> 7 GEMMs + per-row select
  for (int t = 0; t < NTASK; ++t) {
    const v16h* B1 = (const v16h*)(MT1H + t * 4096);
    const v16h* B2 = (const v16h*)(MT2H + t * 4096);
#pragma unroll
    for (int t2 = 0; t2 < 2; ++t2) {
      v8f c1 = {}; v8f c2 = {};
#pragma unroll
      for (int ks = 0; ks < 2; ++ks) {
        int bo = ((ks << 2) | ni[t2]) * 32 + lane;
        c1 = __builtin_amdgcn_wmma_f32_16x16x32_f16(false, aT[t2][ks], false, B1[bo],
                                                    (short)0, c1, false, false);
        c2 = __builtin_amdgcn_wmma_f32_16x16x32_f16(false, aT[t2][ks], false, B2[bo],
                                                    (short)0, c2, false, false);
      }
      store_c(tmp1, 68, mi[t2] * 16, ni[t2] * 16, lane, c1);
      store_c(tmp2, 68, mi[t2] * 16, ni[t2] * 16, lane, c2);
    }
    __syncthreads();
    for (int p = tid; p < 64 * 64; p += 256) {
      int r = p >> 6, e = p & 63;
      if (tidsS[r] == t) {
        embhA[afrag_index(r, e)] = (_Float16)(tmp1[r * 68 + e] + BT1[t * 64 + e]);
        qry[r * 66 + e] = tmp2[r * 68 + e] + BT2[t * 64 + e];
      }
    }
    __syncthreads();
  }

  // actor / critic hidden layers (shared f16 weights) via WMMA
  const v16h* embhAF = (const v16h*)embhA;
  const v16h* BA = (const v16h*)AW1H;
  const v16h* BC = (const v16h*)CW1H;
#pragma unroll
  for (int t2 = 0; t2 < 2; ++t2) {
    v8f c1 = {}; v8f c2 = {};
#pragma unroll
    for (int ks = 0; ks < 2; ++ks) {
      v16h a = embhAF[(mi[t2] * 2 + ks) * 32 + lane];
      int bo = ((ks << 2) | ni[t2]) * 32 + lane;
      c1 = __builtin_amdgcn_wmma_f32_16x16x32_f16(false, a, false, BA[bo],
                                                  (short)0, c1, false, false);
      c2 = __builtin_amdgcn_wmma_f32_16x16x32_f16(false, a, false, BC[bo],
                                                  (short)0, c2, false, false);
    }
    store_c(tmp1, 68, mi[t2] * 16, ni[t2] * 16, lane, c1);
    store_c(tmp2, 68, mi[t2] * 16, ni[t2] * 16, lane, c2);
  }
  __syncthreads();
  for (int p = tid; p < 64 * 64; p += 256) {
    int r = p >> 6, e = p & 63;
    tmp1[r * 68 + e] = tanhf(tmp1[r * 68 + e] + ab1[e]);
    tmp2[r * 68 + e] = tanhf(tmp2[r * 68 + e] + cb1[e]);
  }
  __syncthreads();

  // logits + log_softmax + value (one thread per row)
  if (tid < 64) {
    const float* h1 = tmp1 + tid * 68;
    float lg[NACT];
#pragma unroll
    for (int a = 0; a < NACT; ++a) {
      float s = ab2[a];
      for (int d = 0; d < 64; ++d) s += h1[d] * aw2[d * NACT + a];
      lg[a] = s;
    }
    float mx = lg[0];
#pragma unroll
    for (int a = 1; a < NACT; ++a) mx = fmaxf(mx, lg[a]);
    float se = 0.0f;
#pragma unroll
    for (int a = 0; a < NACT; ++a) se += expf(lg[a] - mx);
    float lse = mx + logf(se);
#pragma unroll
    for (int a = 0; a < NACT; ++a)
      out[(size_t)(row0 + tid) * NACT + a] = lg[a] - lse;

    const float* h2 = tmp2 + tid * 68;
    float v = cb2[0];
    for (int d = 0; d < 64; ++d) v += h2[d] * cw2[d];
    out[(size_t)BATCH * 7 + row0 + tid] = v;
  }
  __syncthreads();

  // attention scores (reuse tmp1, stride 52)
  for (int p = tid; p < 64 * NMEM; p += 256) {
    int r = p / NMEM, n = p - r * NMEM;
    const float* q = qry + r * 66;
    const float* k = KEYV + n * 64;
    float s = 0.0f;
    for (int d = 0; d < 64; ++d) s += q[d] * k[d];
    tmp1[r * 52 + n] = s;
  }
  __syncthreads();
  if (tid < 64) {
    const float* s = tmp1 + tid * 52;
    float mx = s[0];
    for (int n = 1; n < NMEM; ++n) mx = fmaxf(mx, s[n]);
    float Z = 0.0f, wv = 0.0f;
    for (int n = 0; n < NMEM; ++n) {
      float e = expf(s[n] - mx);
      Z += e; wv += e * mem_value[n];
    }
    out[(size_t)BATCH * 8 + row0 + tid] = wv / Z;   // memory_value
    out[(size_t)BATCH * 9 + row0 + tid] = 1.0f / Z; // lam (max of softmax)
  }
}

// ---------------- host-side launcher ----------------
extern "C" void kernel_launch(void* const* d_in, const int* in_sizes, int n_in,
                              void* d_out, int out_size, void* d_ws, size_t ws_size,
                              hipStream_t stream) {
  (void)in_sizes; (void)n_in; (void)out_size; (void)ws_size;
  const float* image     = (const float*)d_in[0];
  const int*   text_ids  = (const int*)d_in[1];
  const float* mem_state = (const float*)d_in[2];
  const float* mem_value = (const float*)d_in[3];
  const int*   mem_tids  = (const int*)d_in[4];
  const float* te1       = (const float*)d_in[5];
  const float* te2       = (const float*)d_in[6];
  const float* Wm        = (const float*)d_in[7];
  const float* bm        = (const float*)d_in[8];
  const float* Wb        = (const float*)d_in[9];
  const float* bb0       = (const float*)d_in[10];
  const float* c1w       = (const float*)d_in[11];
  const float* c1b       = (const float*)d_in[12];
  const float* c2w       = (const float*)d_in[13];
  const float* c2b       = (const float*)d_in[14];
  const float* c3w       = (const float*)d_in[15];
  const float* c3b       = (const float*)d_in[16];
  const float* aw1       = (const float*)d_in[17];
  const float* ab1       = (const float*)d_in[18];
  const float* aw2       = (const float*)d_in[19];
  const float* ab2       = (const float*)d_in[20];
  const float* cw1       = (const float*)d_in[21];
  const float* cb1       = (const float*)d_in[22];
  const float* cw2       = (const float*)d_in[23];
  const float* cb2       = (const float*)d_in[24];

  char* ws = (char*)d_ws;
  _Float16* MT1H = (_Float16*)(ws + WS_MT1H);
  _Float16* MT2H = (_Float16*)(ws + WS_MT2H);
  float*    BT1  = (float*)(ws + WS_BT1);
  float*    BT2  = (float*)(ws + WS_BT2);
  _Float16* AW1H = (_Float16*)(ws + WS_AW1H);
  _Float16* CW1H = (_Float16*)(ws + WS_CW1H);
  float*    KEYV = (float*)(ws + WS_KEYV);
  float*    out  = (float*)d_out;

  acm_precompute<<<8, 256, 0, stream>>>(te1, te2, Wm, bm, Wb, bb0, aw1, cw1,
                                        MT1H, MT2H, BT1, BT2, AW1H, CW1H);
  acm_memory<<<1, 256, SM_MEM_FLOATS * sizeof(float), stream>>>(
      mem_state, mem_tids, c1w, c1b, c2w, c2b, c3w, c3b, MT2H, BT2, KEYV);
  acm_main<<<BATCH / 64, 256, SM_MAIN_FLOATS * sizeof(float), stream>>>(
      image, text_ids, mem_value, c1w, c1b, c2w, c2b, c3w, c3b,
      aw2, ab2, ab1, cb1, cw2, cb2, MT1H, MT2H, BT1, BT2, AW1H, CW1H, KEYV, out);
}
